// FeatureNet_89386859365071
// MI455X (gfx1250) — matrix operand order repaired and checked by reading, not compiled, not run on hardware
//
#include <hip/hip_runtime.h>
#include <hip/hip_bf16.h>

// ---------------------------------------------------------------------------
// FeatureNet on MI455X (gfx1250, wave32):
//   Kernel 1: per-lane KNN (k=8) with all vertices staged in LDS.
//   Kernel 2: fused 4-layer MLP via v_wmma_f32_16x16x32_f16, activations kept
//             entirely in LDS, + inverse-distance weighted neighbor reduce.
//   This rev: fully-unrolled column blocks + double-buffered B fragments so
//   ds_load traffic overlaps the WMMA pipe; single-base immediate-offset
//   epilogue stores; non-canonicalizing ReLU.
// ---------------------------------------------------------------------------

typedef __attribute__((ext_vector_type(16))) _Float16 v16h;
typedef __attribute__((ext_vector_type(8)))  float    v8f;

union Frag16 {
    v16h h;
    unsigned int u[8];
};

// ------------------------------ KNN kernel ---------------------------------
// One query per lane. Vertices (SoA) in dynamic LDS; all lanes iterate the
// same vertex j in lockstep -> LDS broadcast, no bank conflicts.
__global__ __launch_bounds__(256) void knn_kernel(
    const float* __restrict__ vertex, const float* __restrict__ qpts,
    int* __restrict__ kidx, float* __restrict__ kw, int N, int M)
{
    extern __shared__ char smem_raw[];
    float* vx = (float*)smem_raw;
    float* vy = vx + N;
    float* vz = vy + N;

    const int tid = threadIdx.x;
    for (int j = tid; j < N; j += blockDim.x) {
        vx[j] = vertex[3 * j + 0];
        vy[j] = vertex[3 * j + 1];
        vz[j] = vertex[3 * j + 2];
    }
    __syncthreads();

    const int q = blockIdx.x * blockDim.x + tid;
    const bool valid = q < M;
    const int qc = valid ? q : 0;
    const float qx = qpts[qc * 3 + 0];
    const float qy = qpts[qc * 3 + 1];
    const float qz = qpts[qc * 3 + 2];

    float bd[8];
    int   bi[8];
#pragma unroll
    for (int t = 0; t < 8; ++t) { bd[t] = 3.4e38f; bi[t] = 0; }

    for (int j = 0; j < N; ++j) {
        const float dx = qx - vx[j];
        const float dy = qy - vy[j];
        const float dz = qz - vz[j];
        const float d2 = dx * dx + dy * dy + dz * dz;
        if (d2 < bd[7]) {                     // insert + single bubble-up pass
            bd[7] = d2; bi[7] = j;
#pragma unroll
            for (int t = 7; t > 0; --t) {
                if (bd[t] < bd[t - 1]) {
                    float td = bd[t]; bd[t] = bd[t - 1]; bd[t - 1] = td;
                    int   ti = bi[t]; bi[t] = bi[t - 1]; bi[t - 1] = ti;
                }
            }
        }
    }

    if (valid) {
        float inv[8], s = 0.f;
#pragma unroll
        for (int t = 0; t < 8; ++t) {
            const float dd = sqrtf(fmaxf(bd[t], 0.f)) + 1e-9f;
            inv[t] = 1.f / dd;
            s += inv[t];
        }
        const float rs = 1.f / s;
#pragma unroll
        for (int t = 0; t < 8; ++t) {
            kidx[q * 8 + t] = bi[t];
            kw  [q * 8 + t] = inv[t] * rs;
        }
    }
}

// ------------------------- Fused MLP kernel --------------------------------
// LDS layout (bytes), total 232192 (< 320KB WGP LDS):
static constexpr int OFF_W0T  = 0;        // [128n][96k]  f16  24576
static constexpr int OFF_W1T  = 24576;    // [128n][128k] f16  32768
static constexpr int OFF_W2T  = 57344;    // [128n][128k] f16  32768
static constexpr int OFF_W3T  = 90112;    // [ 64n][128k] f16  16384
static constexpr int OFF_BIAS = 106496;   // 448 f32            1792
static constexpr int OFF_X    = 108288;   // [128r][96k]  f16  24576
static constexpr int OFF_HA   = 132864;   // [128r][128]  f16  32768
static constexpr int OFF_HB   = 165632;   // [128r][128]  f16  32768
static constexpr int OFF_Y    = 198400;   // [128r][64]   f32  32768
static constexpr int OFF_IDX  = 231168;   // 128 i32            512
static constexpr int OFF_WQ   = 231680;   // 128 f32            512
static constexpr int SMEM_MLP = 232192;

// One wave computes a 16-row block (rb) against CB 16-col blocks.
// A frag (16x32 f16): m = rb*16 + (lane&15), half = lane>>4,
//   VGPR v in 0..3 -> K = kb*32 + 8*half + 2v{,+1};  v in 4..7 -> +16.
// B frag (32x16 f16): n = cb*16 + (lane&15), same K mapping as A frag.
// C/D (16x16 f32): VGPR v -> M = v + 8*half, N = lane&15.
template <int KSTEPS, int CB, bool RELU, bool F16OUT>
__device__ __forceinline__ void gemm_layer(
    const _Float16* __restrict__ src, int ldS,
    const _Float16* __restrict__ wt,  int ldK,
    const float* __restrict__ bias,
    void* __restrict__ dstv, int ldD,
    int lane, int rb)
{
    const int l15  = lane & 15;
    const int half = lane >> 4;
    const int m    = rb * 16 + l15;

    // Preload all A fragments for this 16-row block (shared across CB blocks).
    Frag16 a[KSTEPS];
#pragma unroll
    for (int kb = 0; kb < KSTEPS; ++kb) {
        const _Float16* ap = src + m * ldS + kb * 32 + 8 * half;
#pragma unroll
        for (int v = 0; v < 4; ++v) a[kb].u[v]     = *(const unsigned int*)(ap + 2 * v);
#pragma unroll
        for (int v = 0; v < 4; ++v) a[kb].u[4 + v] = *(const unsigned int*)(ap + 16 + 2 * v);
    }

#pragma unroll
    for (int cb = 0; cb < CB; ++cb) {
        const int n = cb * 16 + l15;
        const float bv = bias[n];
        v8f acc = {bv, bv, bv, bv, bv, bv, bv, bv};

        const _Float16* bbase = wt + n * ldK + 8 * half;

        // Double-buffered B fragments: loads for kb+1 issue under wmma(kb).
        Frag16 bf[2];
#pragma unroll
        for (int v = 0; v < 8; ++v)
            bf[0].u[v] = *(const unsigned int*)(bbase + 2 * v + (v >> 2) * 8);

#pragma unroll
        for (int kb = 0; kb < KSTEPS; ++kb) {
            if (kb + 1 < KSTEPS) {
#pragma unroll
                for (int v = 0; v < 8; ++v)
                    bf[(kb + 1) & 1].u[v] =
                        *(const unsigned int*)(bbase + (kb + 1) * 32 + 2 * v + (v >> 2) * 8);
            }
            acc = __builtin_amdgcn_wmma_f32_16x16x32_f16(
                false, a[kb].h, false, bf[kb & 1].h, (short)0, acc, false, false);
        }

        // Epilogue: one base pointer, 8 stores at immediate offsets v*ldD.
        if constexpr (F16OUT) {
            _Float16* dstp = (_Float16*)dstv + (rb * 16 + 8 * half) * ldD + n;
#pragma unroll
            for (int v = 0; v < 8; ++v) {
                float x = acc[v];
                if constexpr (RELU) x = (x > 0.f) ? x : 0.f;
                dstp[v * ldD] = (_Float16)x;
            }
        } else {
            float* dstp = (float*)dstv + (rb * 16 + 8 * half) * ldD + n;
#pragma unroll
            for (int v = 0; v < 8; ++v) {
                float x = acc[v];
                if constexpr (RELU) x = (x > 0.f) ? x : 0.f;
                dstp[v * ldD] = x;
            }
        }
    }
}

__global__ __launch_bounds__(256) void mlp_kernel(
    const float* __restrict__ vertex, const float* __restrict__ vfeat,
    const float* __restrict__ qpts,
    const float* __restrict__ W0, const float* __restrict__ b0,
    const float* __restrict__ W1, const float* __restrict__ b1,
    const float* __restrict__ W2, const float* __restrict__ b2,
    const float* __restrict__ W3, const float* __restrict__ b3,
    const int* __restrict__ kidx, const float* __restrict__ kw,
    float* __restrict__ out, int M)
{
    extern __shared__ char smem_raw[];
    _Float16* w0t  = (_Float16*)(smem_raw + OFF_W0T);
    _Float16* w1t  = (_Float16*)(smem_raw + OFF_W1T);
    _Float16* w2t  = (_Float16*)(smem_raw + OFF_W2T);
    _Float16* w3t  = (_Float16*)(smem_raw + OFF_W3T);
    float*    bias = (float*)   (smem_raw + OFF_BIAS);
    _Float16* X    = (_Float16*)(smem_raw + OFF_X);
    _Float16* HA   = (_Float16*)(smem_raw + OFF_HA);
    _Float16* HB   = (_Float16*)(smem_raw + OFF_HB);
    float*    Yf   = (float*)   (smem_raw + OFF_Y);
    int*      idxs = (int*)     (smem_raw + OFF_IDX);
    float*    wqs  = (float*)   (smem_raw + OFF_WQ);

    const int tid  = threadIdx.x;
    const int lane = tid & 31;
    const int rb   = tid >> 5;           // wave id = 16-row block
    const int q0   = blockIdx.x * 16;    // 16 queries * 8 neighbors = 128 rows
    const int row0 = q0 * 8;

    // ---- stage weights (f16, transposed [N][K]) and biases into LDS ----
    for (int i = tid; i < 128 * 96; i += 256) {
        const int n = i / 96, k = i - n * 96;
        w0t[i] = (k < 67) ? (_Float16)W0[k * 128 + n] : (_Float16)0.f;
    }
    for (int i = tid; i < 128 * 128; i += 256) {
        const int n = i >> 7, k = i & 127;
        w1t[i] = (_Float16)W1[k * 128 + n];
        w2t[i] = (_Float16)W2[k * 128 + n];
    }
    for (int i = tid; i < 64 * 128; i += 256) {
        const int n = i >> 7, k = i & 127;
        w3t[i] = (_Float16)W3[k * 64 + n];
    }
    for (int i = tid; i < 448; i += 256) {
        float v;
        if      (i < 128) v = b0[i];
        else if (i < 256) v = b1[i - 128];
        else if (i < 384) v = b2[i - 256];
        else              v = b3[i - 384];
        bias[i] = v;
    }
    if (tid < 128) {
        const int r = row0 + tid;
        const bool ok = r < M * 8;
        idxs[tid] = ok ? kidx[r] : 0;
        wqs [tid] = ok ? kw[r]   : 0.f;
    }
    __syncthreads();

    // ---- build X tile [128 rows x 96 (=64 feat + 3 vec + pad)] in f16 ----
    for (int i = tid; i < 128 * 96; i += 256) {
        const int r = i / 96, c = i - r * 96;
        const int vi = idxs[r];
        float val;
        if (c < 64) {
            val = vfeat[vi * 64 + c];
        } else if (c < 67) {
            const int ax = c - 64;
            int qg = q0 + (r >> 3);
            qg = qg < M ? qg : 0;
            val = qpts[qg * 3 + ax] - vertex[vi * 3 + ax];
        } else {
            val = 0.f;
        }
        X[i] = (_Float16)val;
    }
    __syncthreads();

    // ---- fused MLP: 67->128->128->128->64, all activations in LDS ----
    gemm_layer<3, 8, true,  true >(X,  96,  w0t, 96,  bias,       HA, 128, lane, rb);
    __syncthreads();
    gemm_layer<4, 8, true,  true >(HA, 128, w1t, 128, bias + 128, HB, 128, lane, rb);
    __syncthreads();
    gemm_layer<4, 8, true,  true >(HB, 128, w2t, 128, bias + 256, HA, 128, lane, rb);
    __syncthreads();
    gemm_layer<4, 4, false, false>(HA, 128, w3t, 128, bias + 384, Yf, 64,  lane, rb);
    __syncthreads();

    // ---- inverse-distance weighted reduction over 8 neighbor rows ----
    for (int j = tid; j < 16 * 64; j += 256) {
        const int ql = j >> 6, d = j & 63;
        float s = 0.f;
#pragma unroll
        for (int k = 0; k < 8; ++k)
            s += wqs[ql * 8 + k] * Yf[(ql * 8 + k) * 64 + d];
        const int qg = q0 + ql;
        if (qg < M) out[qg * 64 + d] = s;
    }
}

// ------------------------------- launcher ----------------------------------
extern "C" void kernel_launch(void* const* d_in, const int* in_sizes, int n_in,
                              void* d_out, int out_size, void* d_ws, size_t ws_size,
                              hipStream_t stream) {
    const float* vertex = (const float*)d_in[0];
    const float* vfeat  = (const float*)d_in[1];
    const float* qpts   = (const float*)d_in[2];
    const float* W0 = (const float*)d_in[3];
    const float* b0 = (const float*)d_in[4];
    const float* W1 = (const float*)d_in[5];
    const float* b1 = (const float*)d_in[6];
    const float* W2 = (const float*)d_in[7];
    const float* b2 = (const float*)d_in[8];
    const float* W3 = (const float*)d_in[9];
    const float* b3 = (const float*)d_in[10];

    const int N = in_sizes[0] / 3;   // 6890 vertices
    const int M = in_sizes[2] / 3;   // 32768 queries

    int*   kidx = (int*)d_ws;                                          // [M*8] i32
    float* kw   = (float*)((char*)d_ws + (size_t)M * 8 * sizeof(int)); // [M*8] f32

    const int    knnBlocks = (M + 255) / 256;
    const size_t knnSmem   = (size_t)3 * N * sizeof(float);            // ~83 KB
    knn_kernel<<<knnBlocks, 256, knnSmem, stream>>>(vertex, qpts, kidx, kw, N, M);

    const int mlpBlocks = (M + 15) / 16;                               // 2048
    mlp_kernel<<<mlpBlocks, 256, SMEM_MLP, stream>>>(
        vertex, vfeat, qpts, W0, b0, W1, b1, W2, b2, W3, b3,
        kidx, kw, (float*)d_out, M);
}
